// GlobalTransformerLayer_22016002359712
// MI455X (gfx1250) — matrix operand order repaired and checked by reading, not compiled, not run on hardware
//
#include <hip/hip_runtime.h>
#include <hip/hip_bf16.h>

// MI455X / gfx1250 Graphormer-style transformer layer.
// - All matmuls via v_wmma_f32_16x16x32_f16 (wave32).
// - GEMM block tiles staged into LDS with the Tensor Data Mover
//   (tensor_load_to_lds + s_wait_tensorcnt double buffering, padded LDS rows
//   to avoid bank conflicts).
// - Flash attention with online softmax; P goes C-layout -> LDS -> A-layout.

typedef _Float16 f16;
typedef __attribute__((ext_vector_type(16))) _Float16 v16h;
typedef __attribute__((ext_vector_type(8)))  _Float16 v8h;
typedef __attribute__((ext_vector_type(8)))  float    v8f;
typedef __attribute__((ext_vector_type(4)))  unsigned int u32x4;
typedef __attribute__((ext_vector_type(8)))  int i32x8;
typedef __attribute__((ext_vector_type(4)))  int i32x4;

#define DEV __device__ __forceinline__

// ---------------------------------------------------------------------------
// WMMA fragment loaders (wave32, 16x16x32 f16 -> f32)
// A (16x32, MxK): lane m = lane&15; element e: v=e>>1:
//   K = 2v + (e&1) + 16*(lane>=16)
// B (32x16, KxN): lane n = lane&15; e -> K = e + 16*(lane>=16)
// C/D (16x16): lane col n = lane&15; vgpr r -> row = r + 8*(lane>=16)
// ---------------------------------------------------------------------------
DEV v16h loadA16x32(const f16* __restrict__ base, int ld, int lane) {
  const int m = lane & 15;
  const int h8 = (lane >> 4) << 4;
  const f16* r = base + (size_t)m * ld + h8;
  v16h a;
#pragma unroll
  for (int v = 0; v < 8; ++v) {
    const int Kb = 2 * v;   // 0,2,4,...,14
    a[2 * v]     = r[Kb];
    a[2 * v + 1] = r[Kb + 1];
  }
  return a;
}

// B element (K,n) stored at base[n*ld + K]
DEV v16h loadB_nmajor(const f16* __restrict__ base, int ld, int lane) {
  const int n  = lane & 15;
  const int kh = (lane >> 4) << 4;
  const f16* r = base + (size_t)n * ld + kh;
  v16h b;
#pragma unroll
  for (int e = 0; e < 16; ++e) b[e] = r[e];
  return b;
}

// B element (K,n) stored at base[K*ld + n]  (V matrix: rows = keys = K dim)
DEV v16h loadB_kmajor(const f16* __restrict__ base, int ld, int lane) {
  const int n  = lane & 15;
  const int kh = (lane >> 4) << 4;
  v16h b;
#pragma unroll
  for (int e = 0; e < 16; ++e) b[e] = base[(size_t)(kh + e) * ld + n];
  return b;
}

DEV v8f wmma32(v16h a, v16h b, v8f c) {
  return __builtin_amdgcn_wmma_f32_16x16x32_f16(false, a, false, b,
                                                (short)0, c, false, false);
}

// ---------------------------------------------------------------------------
// Tensor Data Mover: async 2D tile load (rows x 32 f16) global -> LDS with
// 1-dword pad every 16 dwords (LDS row pitch = 34 f16 = 68 B, bank-friendly).
// D# per CDNA5 ISA.
// ---------------------------------------------------------------------------
#define PADLD 34

DEV void tdm_load_tile(const f16* ldsDst, const f16* gsrc, int ld, int rows) {
  const unsigned lds_off = (unsigned)(uintptr_t)ldsDst;   // low 32b = LDS offset
  const unsigned long long ga = (unsigned long long)(uintptr_t)gsrc;
  u32x4 g0;
  g0[0] = 1u;                                             // count=1 (valid user D#)
  g0[1] = lds_off;                                        // lds_addr
  g0[2] = (unsigned)(ga & 0xffffffffu);                   // global_addr[31:0]
  g0[3] = (unsigned)((ga >> 32) & 0x01ffffffu) | (2u << 30); // addr[56:32] | type=2
  i32x8 g1;
  g1[0] = (int)((1u << 16) | (1u << 20) | (3u << 22));    // data_size=2B, pad_en,
                                                          // pad_interval=16dw, pad=1dw
  g1[1] = (int)(32u << 16);                               // tensor_dim0 = 32
  g1[2] = (int)(((unsigned)rows & 0xffffu) << 16);        // tensor_dim1 = rows
  g1[3] = (int)(32u << 16);                               // tile_dim0 = 32
  g1[4] = (int)rows;                                      // tile_dim1 = rows (dim2=0)
  g1[5] = (int)ld;                                        // tensor_dim0_stride lo
  g1[6] = 0;
  g1[7] = 0;
  i32x4 z4 = {0, 0, 0, 0};                                // groups 2/3 unused (2D)
#if defined(__clang_major__) && (__clang_major__ >= 23)
  i32x8 z8 = {0, 0, 0, 0, 0, 0, 0, 0};
  __builtin_amdgcn_tensor_load_to_lds(g0, g1, z4, z4, z8, 0);
#else
  __builtin_amdgcn_tensor_load_to_lds(g0, g1, z4, z4, 0);
#endif
}

// ---------------------------------------------------------------------------
// fp32 -> f16 conversion
// ---------------------------------------------------------------------------
__global__ __launch_bounds__(256) void k_cvt(const float* __restrict__ in,
                                             f16* __restrict__ out, int n) {
  int i = blockIdx.x * 256 + threadIdx.x;
  if (i < n) out[i] = (f16)in[i];
}

// ---------------------------------------------------------------------------
// LayerNorm: one wave per row of 512; wave32 shuffle reductions.
// ---------------------------------------------------------------------------
__global__ __launch_bounds__(256) void k_layernorm(
    const float* __restrict__ x, const float* __restrict__ g,
    const float* __restrict__ bta, float* __restrict__ y32,
    f16* __restrict__ y16) {
  const int row  = blockIdx.x * 8 + (threadIdx.x >> 5);
  const int lane = threadIdx.x & 31;
  const float* xr = x + (size_t)row * 512;
  float vals[16];
  float s = 0.f;
#pragma unroll
  for (int i = 0; i < 16; ++i) { vals[i] = xr[lane + 32 * i]; s += vals[i]; }
#pragma unroll
  for (int m = 16; m >= 1; m >>= 1) s += __shfl_xor(s, m, 32);
  const float mean = s * (1.0f / 512.0f);
  float vs = 0.f;
#pragma unroll
  for (int i = 0; i < 16; ++i) { float d = vals[i] - mean; vs += d * d; }
#pragma unroll
  for (int m = 16; m >= 1; m >>= 1) vs += __shfl_xor(vs, m, 32);
  const float rstd = rsqrtf(vs * (1.0f / 512.0f) + 1e-5f);
#pragma unroll
  for (int i = 0; i < 16; ++i) {
    const int c = lane + 32 * i;
    const float o = (vals[i] - mean) * rstd * g[c] + bta[c];
    if (y32) y32[(size_t)row * 512 + c] = o;
    y16[(size_t)row * 512 + c] = (f16)o;
  }
}

// ---------------------------------------------------------------------------
// Attention bias via WMMA: A[m=pair, K=gaussian]=exp(..), B[K=g, n=h]=rbf_w[h][g]
// ---------------------------------------------------------------------------
__global__ __launch_bounds__(256) void k_bias(
    const int* __restrict__ dm, const float* __restrict__ d3,
    const float* __restrict__ demb, const float* __restrict__ rbfw,
    const float* __restrict__ rbfb, f16* __restrict__ bias) {
  const int lane = threadIdx.x & 31;
  const int wave = threadIdx.x >> 5;
  const int id = blockIdx.x * 8 + wave;   // B*N*(N/16) wave-tiles
  const int jt = id & 31;
  const int i  = (id >> 5) & 511;
  const int b  = id >> 14;
  const int m = lane & 15, half = lane >> 4;
  const int j = (jt << 4) + m;

  const float dist  = d3[((size_t)b * 512 + i) * 512 + j];
  const float step  = 20.0f / 31.0f;
  const float coeff = -0.5f / (step * step);
  v16h a;
#pragma unroll
  for (int v = 0; v < 8; ++v) {
    const int Kb = 2 * v + (half << 4);
    const float d0 = dist - step * (float)Kb;
    const float d1 = dist - step * (float)(Kb + 1);
    a[2 * v]     = (f16)__expf(coeff * d0 * d0);
    a[2 * v + 1] = (f16)__expf(coeff * d1 * d1);
  }
  const int n  = lane & 15;
  const int kh = (lane >> 4) << 4;
  v16h bw;
#pragma unroll
  for (int e = 0; e < 16; ++e) bw[e] = (f16)rbfw[n * 32 + kh + e];

  v8f c = {};
  c = wmma32(a, bw, c);

  const float rb = rbfb[n];
  const int jbase = (jt << 4) + (half << 3);
  v8h stv;
#pragma unroll
  for (int r = 0; r < 8; ++r) {
    const int jj = jbase + r;
    int dc = dm[((size_t)b * 512 + i) * 512 + jj];
    dc = dc < 0 ? 0 : (dc > 127 ? 127 : dc);
    stv[r] = (f16)(c[r] + rb + demb[dc * 16 + n]);
  }
  const size_t ob = (((size_t)b * 16 + n) * 512 + i) * 512 + jbase;
  *(v8h*)(bias + ob) = stv;
}

// ---------------------------------------------------------------------------
// WMMA GEMM with TDM double-buffered LDS staging.
// Block = 256 thr (8 waves); block tile 128x64; wave tile 32x32 (2x2 C-tiles).
// Wave 0 drives the Tensor Data Mover; TENSORcnt + s_barrier for pipelining.
// mode: 0 -> f16 store; 1 -> exact GELU then f16; 2 -> fp32 resid + store.
// ---------------------------------------------------------------------------
DEV void epilogue_tile(v8f c, int row0, int col0, int lane,
                       const float* __restrict__ bias,
                       const float* __restrict__ resid,
                       f16* __restrict__ out16, float* __restrict__ out32,
                       int ldo, int mode) {
  const int n = lane & 15, half = lane >> 4;
  const float bn = bias[col0 + n];
#pragma unroll
  for (int r = 0; r < 8; ++r) {
    const int row = row0 + r + half * 8;
    const size_t idx = (size_t)row * ldo + col0 + n;
    float v = c[r] + bn;
    if (mode == 1) v = 0.5f * v * (1.0f + erff(v * 0.70710678118654752f));
    if (mode == 2) out32[idx] = resid[idx] + v;
    else           out16[idx] = (f16)v;
  }
}

__global__ __launch_bounds__(256) void k_gemm(
    const f16* __restrict__ A, int lda, const f16* __restrict__ W, int ldw,
    const float* __restrict__ bias, const float* __restrict__ resid,
    f16* __restrict__ out16, float* __restrict__ out32, int ldo,
    int K, int mode) {
  __shared__ f16 ldsA[2][128 * PADLD];
  __shared__ f16 ldsB[2][64 * PADLD];
  const int lane = threadIdx.x & 31;
  const int wave = threadIdx.x >> 5;
  const int row0b = blockIdx.y * 128;
  const int col0b = blockIdx.x * 64;
  const int wrow = (wave & 3) * 32;
  const int wcol = (wave >> 2) * 32;
  const f16* Ab = A + (size_t)row0b * lda;
  const f16* Wb = W + (size_t)col0b * ldw;

  if (threadIdx.x < 32) {       // wave 0 drives the TDM (EXEC-independent DMA)
    tdm_load_tile(&ldsA[0][0], Ab, lda, 128);
    tdm_load_tile(&ldsB[0][0], Wb, ldw, 64);
  }
  v8f c00 = {}, c01 = {}, c10 = {}, c11 = {};
  const int nk = K / 32;
  int buf = 0;
  for (int kt = 0; kt < nk; ++kt) {
    if (threadIdx.x < 32) {
      if (kt + 1 < nk) {
        const int k0n = (kt + 1) * 32;
        tdm_load_tile(&ldsA[buf ^ 1][0], Ab + k0n, lda, 128);
        tdm_load_tile(&ldsB[buf ^ 1][0], Wb + k0n, ldw, 64);
        __builtin_amdgcn_s_wait_tensorcnt(2);   // oldest pair (current buf) done
      } else {
        __builtin_amdgcn_s_wait_tensorcnt(0);
      }
    }
    __syncthreads();
    const f16* At = &ldsA[buf][0] + wrow * PADLD;
    const f16* Bt = &ldsB[buf][0] + wcol * PADLD;
    v16h a0 = loadA16x32(At, PADLD, lane);
    v16h a1 = loadA16x32(At + 16 * PADLD, PADLD, lane);
    v16h b0 = loadB_nmajor(Bt, PADLD, lane);
    v16h b1 = loadB_nmajor(Bt + 16 * PADLD, PADLD, lane);
    c00 = wmma32(a0, b0, c00);
    c01 = wmma32(a0, b1, c01);
    c10 = wmma32(a1, b0, c10);
    c11 = wmma32(a1, b1, c11);
    __syncthreads();            // protect buf from next TDM issue
    buf ^= 1;
  }
  epilogue_tile(c00, row0b + wrow,      col0b + wcol,      lane, bias, resid, out16, out32, ldo, mode);
  epilogue_tile(c01, row0b + wrow,      col0b + wcol + 16, lane, bias, resid, out16, out32, ldo, mode);
  epilogue_tile(c10, row0b + wrow + 16, col0b + wcol,      lane, bias, resid, out16, out32, ldo, mode);
  epilogue_tile(c11, row0b + wrow + 16, col0b + wcol + 16, lane, bias, resid, out16, out32, ldo, mode);
}

// ---------------------------------------------------------------------------
// Flash attention. One wave owns a 16-query tile of one (b, h); keys in
// chunks of 32: two score WMMAs (K = dh = 32), bias + scale, online softmax
// (width-16 shfl_xor), P through per-wave LDS tile (s_wait_dscnt), PV WMMAs.
// ---------------------------------------------------------------------------
__global__ __launch_bounds__(128) void k_attn(
    const f16* __restrict__ qkv, const f16* __restrict__ bias,
    f16* __restrict__ ctx) {
  __shared__ f16 ldsP[4][16 * 32];
  const int lane = threadIdx.x & 31;
  const int wave = threadIdx.x >> 5;
  const int id = blockIdx.x;                  // (b, h, qblk)
  const int qb = id & 7;
  const int h  = (id >> 3) & 15;
  const int b  = id >> 7;
  const int q0 = qb * 64 + wave * 16;

  const f16* Qp = qkv + ((size_t)b * 512 + q0) * 1536 + h * 32;
  const f16* Kp = qkv + (size_t)b * 512 * 1536 + 512  + h * 32;
  const f16* Vp = qkv + (size_t)b * 512 * 1536 + 1024 + h * 32;
  const f16* Bp = bias + (((size_t)b * 16 + h) * 512 + q0) * 512;

  const v16h aQ = loadA16x32(Qp, 1536, lane);
  v8f ctx0 = {}, ctx1 = {};
  float rmax[8], rsum[8];
#pragma unroll
  for (int r = 0; r < 8; ++r) { rmax[r] = -3.0e38f; rsum[r] = 0.f; }

  const int n = lane & 15, half = lane >> 4;
  const float scale = 0.17677669529663688f;   // 1/sqrt(32)
  f16* P = &ldsP[wave][0];

  for (int t = 0; t < 16; ++t) {
    const int key0 = t * 32;
    v16h k0f = loadA16x32(Kp + (size_t)key0 * 1536, 1536, lane);
    v16h k1f = loadA16x32(Kp + (size_t)(key0 + 16) * 1536, 1536, lane);
    v8f z = {};
    v8f s0 = wmma32(aQ, k0f, z);
    v8f s1 = wmma32(aQ, k1f, z);

    float p0[8], p1[8];
#pragma unroll
    for (int r = 0; r < 8; ++r) {
      const int qr = r + half * 8;
      const float bv0 = (float)Bp[(size_t)qr * 512 + key0 + n];
      const float bv1 = (float)Bp[(size_t)qr * 512 + key0 + 16 + n];
      s0[r] = s0[r] * scale + bv0;
      s1[r] = s1[r] * scale + bv1;
    }
#pragma unroll
    for (int r = 0; r < 8; ++r) {
      float tm = fmaxf(s0[r], s1[r]);
#pragma unroll
      for (int m2 = 8; m2 >= 1; m2 >>= 1) tm = fmaxf(tm, __shfl_xor(tm, m2, 16));
      const float nm = fmaxf(rmax[r], tm);
      const float corr = __expf(rmax[r] - nm);
      rmax[r] = nm;
      const float e0 = __expf(s0[r] - nm);
      const float e1 = __expf(s1[r] - nm);
      float ps = e0 + e1;
#pragma unroll
      for (int m2 = 8; m2 >= 1; m2 >>= 1) ps += __shfl_xor(ps, m2, 16);
      rsum[r] = rsum[r] * corr + ps;
      ctx0[r] *= corr;
      ctx1[r] *= corr;
      p0[r] = e0;
      p1[r] = e1;
    }
#pragma unroll
    for (int r = 0; r < 8; ++r) {
      const int qr = r + half * 8;
      P[qr * 32 + n]      = (f16)p0[r];
      P[qr * 32 + 16 + n] = (f16)p1[r];
    }
    asm volatile("s_wait_dscnt 0" ::: "memory");
    v16h aP  = loadA16x32(P, 32, lane);
    v16h v0f = loadB_kmajor(Vp + (size_t)key0 * 1536, 1536, lane);
    v16h v1f = loadB_kmajor(Vp + (size_t)key0 * 1536 + 16, 1536, lane);
    ctx0 = wmma32(aP, v0f, ctx0);
    ctx1 = wmma32(aP, v1f, ctx1);
  }
#pragma unroll
  for (int r = 0; r < 8; ++r) {
    const float inv = 1.0f / rsum[r];
    const size_t row = (size_t)b * 512 + q0 + r + half * 8;
    ctx[row * 512 + h * 32 + n]      = (f16)(ctx0[r] * inv);
    ctx[row * 512 + h * 32 + 16 + n] = (f16)(ctx1[r] * inv);
  }
}

// ---------------------------------------------------------------------------
// Host-side orchestration
// ---------------------------------------------------------------------------
extern "C" void kernel_launch(void* const* d_in, const int* in_sizes, int n_in,
                              void* d_out, int out_size, void* d_ws, size_t ws_size,
                              hipStream_t stream) {
  const float* x          = (const float*)d_in[0];
  const int*   dist_mat   = (const int*)  d_in[1];
  const float* dist_3d    = (const float*)d_in[2];
  const float* norm1_g    = (const float*)d_in[3];
  const float* norm1_b    = (const float*)d_in[4];
  const float* in_proj_w  = (const float*)d_in[5];
  const float* in_proj_b  = (const float*)d_in[6];
  const float* out_proj_w = (const float*)d_in[7];
  const float* out_proj_b = (const float*)d_in[8];
  const float* dist_emb   = (const float*)d_in[9];
  const float* rbf_w      = (const float*)d_in[10];
  const float* rbf_b      = (const float*)d_in[11];
  const float* norm2_g    = (const float*)d_in[12];
  const float* norm2_b    = (const float*)d_in[13];
  const float* ffn_w1     = (const float*)d_in[14];
  const float* ffn_b1     = (const float*)d_in[15];
  const float* ffn_w2     = (const float*)d_in[16];
  const float* ffn_b2     = (const float*)d_in[17];
  float* out = (float*)d_out;

  char* ws = (char*)d_ws;
  size_t off = 0;
  auto take = [&](size_t bytes) { char* p = ws + off; off += (bytes + 255) & ~(size_t)255; return p; };
  float* h32    = (float*)take((size_t)4096 * 512 * 4);
  f16*   h16    = (f16*)  take((size_t)4096 * 512 * 2);
  f16*   wqkv16 = (f16*)  take((size_t)1536 * 512 * 2);
  f16*   wout16 = (f16*)  take((size_t)512 * 512 * 2);
  f16*   w1_16  = (f16*)  take((size_t)2048 * 512 * 2);
  f16*   w2_16  = (f16*)  take((size_t)512 * 2048 * 2);
  f16*   qkv16  = (f16*)  take((size_t)4096 * 1536 * 2);
  f16*   ctx16  = (f16*)  take((size_t)4096 * 512 * 2);
  float* h2_32  = (float*)take((size_t)4096 * 512 * 4);
  f16*   h2n16  = (f16*)  take((size_t)4096 * 512 * 2);
  f16*   ff16   = (f16*)  take((size_t)4096 * 2048 * 2);
  f16*   bias16 = (f16*)  take((size_t)8 * 16 * 512 * 512 * 2);

  // weights -> f16
  k_cvt<<<1536 * 512 / 256, 256, 0, stream>>>(in_proj_w, wqkv16, 1536 * 512);
  k_cvt<<<512 * 512 / 256,  256, 0, stream>>>(out_proj_w, wout16, 512 * 512);
  k_cvt<<<2048 * 512 / 256, 256, 0, stream>>>(ffn_w1, w1_16, 2048 * 512);
  k_cvt<<<512 * 2048 / 256, 256, 0, stream>>>(ffn_w2, w2_16, 512 * 2048);

  // LN1 (fp32 kept: reference residual uses the *normed* activations)
  k_layernorm<<<4096 / 8, 256, 0, stream>>>(x, norm1_g, norm1_b, h32, h16);

  // attention bias [B,H,N,N] in f16
  k_bias<<<8 * 512 * 32 / 8, 256, 0, stream>>>(dist_mat, dist_3d, dist_emb,
                                               rbf_w, rbf_b, bias16);

  // QKV: [4096,512] @ [512,1536]
  k_gemm<<<dim3(1536 / 64, 4096 / 128), 256, 0, stream>>>(
      h16, 512, wqkv16, 512, in_proj_b, nullptr, qkv16, nullptr, 1536, 512, 0);

  // flash attention -> ctx f16 [4096,512]
  k_attn<<<8 * 16 * 8, 128, 0, stream>>>(qkv16, bias16, ctx16);

  // out_proj + residual (fp32): h2 = h + ctx @ Wout^T + b
  k_gemm<<<dim3(512 / 64, 4096 / 128), 256, 0, stream>>>(
      ctx16, 512, wout16, 512, out_proj_b, h32, nullptr, h2_32, 512, 512, 2);

  // LN2 -> f16
  k_layernorm<<<4096 / 8, 256, 0, stream>>>(h2_32, norm2_g, norm2_b, nullptr, h2n16);

  // FFN1 + exact GELU: [4096,512] @ [512,2048]
  k_gemm<<<dim3(2048 / 64, 4096 / 128), 256, 0, stream>>>(
      h2n16, 512, w1_16, 512, ffn_b1, nullptr, ff16, nullptr, 2048, 512, 1);

  // FFN2 + residual -> d_out (fp32)
  k_gemm<<<dim3(512 / 64, 4096 / 128), 256, 0, stream>>>(
      ff16, 2048, w2_16, 2048, ffn_b2, h2_32, nullptr, out, 512, 2048, 2);

  (void)in_sizes; (void)n_in; (void)out_size; (void)ws_size;
}
